// WLGNN_5549097746502
// MI455X (gfx1250) — matrix operand order repaired and checked by reading, not compiled, not run on hardware
//
#include <hip/hip_runtime.h>
#include <hip/hip_bf16.h>

#define P_TOT   60000
#define E2_CNT  960000
#define EI_CNT  30000
#define NL_CNT  60000
#define NNODE   1024
#define NFEAT   128
#define L1F     32
#define L2F     24
#define LN_EPS  1e-5f

typedef __attribute__((ext_vector_type(16))) _Float16 v16h;
typedef __attribute__((ext_vector_type(8)))  _Float16 v8h;
typedef __attribute__((ext_vector_type(8)))  float    v8f;

__device__ __forceinline__ float wave_sum32(float v) {
#pragma unroll
  for (int o = 16; o > 0; o >>= 1) v += __shfl_xor(v, o, 32);
  return v;
}

// ---------------- small utility kernels ----------------

__global__ void k_cast_f16(const float* __restrict__ s, _Float16* __restrict__ d, int n) {
  int i = blockIdx.x * blockDim.x + threadIdx.x;
  if (i < n) d[i] = (_Float16)s[i];
}

__global__ void k_gather_cast(const float* __restrict__ x, const int* __restrict__ mask,
                              _Float16* __restrict__ d, int rows) {
  int i = blockIdx.x * blockDim.x + threadIdx.x;
  int n = rows * NFEAT;
  if (i < n) {
    int r = mask[i >> 7];
    d[i] = (_Float16)x[r * NFEAT + (i & 127)];
  }
}

__global__ void k_deg(const int* __restrict__ col, int* __restrict__ deg, int ne) {
  int i = blockIdx.x * blockDim.x + threadIdx.x;
  if (i < ne) atomicAdd(&deg[col[i]], 1);
}

__global__ void k_dinv(const int* __restrict__ deg, float* __restrict__ dinv, int n) {
  int i = blockIdx.x * blockDim.x + threadIdx.x;
  if (i < n) dinv[i] = rsqrtf((float)(deg[i] + 1));  // +1 = self loop
}

// ---------------- B-packing: f32 weights -> f16 in exact WMMA lane order ----------------
// Bp[((tn*K32 + kb32)*32 + lane)*16 + e] = B[(kb32*32 + (e<8 ? 8*(lane>>4)+e : 16+8*(lane>>4)+e-8))*N + tn*16+(lane&15)]
// so each GEMM lane reads one contiguous, 32B-aligned v16h per k-step. Zero-pads cols >= N.
__global__ void k_pack_b(const float* __restrict__ B, _Float16* __restrict__ Bp, int N, int K) {
  int idx = blockIdx.x * blockDim.x + threadIdx.x;
  int K32 = K >> 5;
  int NT = (N + 15) >> 4;
  int total = NT * K32 * 32;
  if (idx >= total) return;
  int lane = idx & 31;
  int rest = idx >> 5;
  int kb32 = rest % K32;
  int tn = rest / K32;
  int hi = lane >> 4;
  int n = tn * 16 + (lane & 15);
  bool nok = (n < N);
#pragma unroll
  for (int e = 0; e < 16; ++e) {
    int kk = kb32 * 32 + (e < 8 ? 8 * hi + e : 16 + 8 * hi + (e - 8));
    Bp[idx * 16 + e] = nok ? (_Float16)B[kk * N + n] : (_Float16)0.f;
  }
}

// ---------------- WMMA f16 GEMM:  C[MxN] = A[MxK] * Bpacked  (f32 accum) ----------------
// M multiple of 16; K in {32,128} (template, fully unrolled); N arbitrary (store-guarded).
template <int K>
__global__ void k_gemm_wmma(const _Float16* __restrict__ A, const _Float16* __restrict__ Bp,
                            float* __restrict__ C, int M, int N) {
  constexpr int K32 = K / 32;
  const int lane = threadIdx.x & 31;
  const int wid  = threadIdx.x >> 5;
  const int MT = M >> 4;
  const int NT = (N + 15) >> 4;
  int tile = blockIdx.x * 4 + wid;           // blockDim.x == 128 (4 waves)
  if (tile >= MT * NT) return;
  const int tn = tile % NT;
  const int tm = tile / NT;
  const int hi   = lane >> 4;
  const int ncol = tn * 16 + (lane & 15);

  const _Float16* ap = A + (tm * 16 + (lane & 15)) * K + 8 * hi;
  const _Float16* bp = Bp + tn * K32 * 512 + lane * 16;

  v8f c = {};
#pragma unroll
  for (int kb32 = 0; kb32 < K32; ++kb32) {
    v8h lo = *(const v8h*)(ap + kb32 * 32);
    v8h hh = *(const v8h*)(ap + kb32 * 32 + 16);
    v16h b = *(const v16h*)(bp + kb32 * 512);
    if (kb32 + 1 < K32) __builtin_prefetch(ap + kb32 * 32 + 32, 0, 0);
    v16h a;
#pragma unroll
    for (int i = 0; i < 8; ++i) { a[i] = lo[i]; a[8 + i] = hh[i]; }
    c = __builtin_amdgcn_wmma_f32_16x16x32_f16(false, a, false, b, (short)0, c, false, false);
  }
  if (ncol < N) {
#pragma unroll
    for (int r = 0; r < 8; ++r) {
      int row = tm * 16 + r + 8 * hi;      // C layout: VGPR r -> M = r (+8 for hi lanes)
      C[row * N + ncol] = c[r];
    }
  }
}

// ---------------- GCN scatter: acc[col] += t[row] * dinv[row]*dinv[col] ----------------
__global__ void k_scatter(const int* __restrict__ row, const int* __restrict__ col,
                          const float* __restrict__ dinv, const float* __restrict__ t,
                          float* __restrict__ acc, int ne) {
  int gid = blockIdx.x * blockDim.x + threadIdx.x;
  if (gid >= ne * 32) return;
  int e = gid >> 5, ch = gid & 31;
  int r = row[e], cl = col[e];
  float nrm = dinv[r] * dinv[cl];
  atomicAdd(&acc[cl * 32 + ch], t[r * 32 + ch] * nrm);
}

// ---------------- GCN epilogue: self-loop + bias + LN + ReLU for both branches, sum ----------------
__global__ void k_gcn_epi(const float* __restrict__ acc0, const float* __restrict__ t0,
                          const float* __restrict__ dinv0, const float* __restrict__ b0,
                          const float* __restrict__ acc1, const float* __restrict__ t1,
                          const float* __restrict__ dinv1, const float* __restrict__ b1,
                          float* __restrict__ hout, _Float16* __restrict__ houth, int n) {
  int lane = threadIdx.x & 31;
  int node = blockIdx.x * (blockDim.x >> 5) + (threadIdx.x >> 5);
  if (node >= n) return;
  int idx = node * 32 + lane;
  float d0 = dinv0[node], d1 = dinv1[node];
  float va = acc0[idx] + t0[idx] * d0 * d0 + b0[lane];
  float vb = acc1[idx] + t1[idx] * d1 * d1 + b1[lane];

  float mu = wave_sum32(va) * (1.f / 32.f);
  float dv = va - mu;
  float var = wave_sum32(dv * dv) * (1.f / 32.f);
  va = fmaxf(dv * rsqrtf(var + LN_EPS), 0.f);

  mu = wave_sum32(vb) * (1.f / 32.f);
  dv = vb - mu;
  var = wave_sum32(dv * dv) * (1.f / 32.f);
  vb = fmaxf(dv * rsqrtf(var + LN_EPS), 0.f);

  float h = va + vb;
  hout[idx] = h;
  if (houth) houth[idx] = (_Float16)h;
}

// ---------------- MLP epilogue: in-place bias + LN(24) + ReLU ----------------
__global__ void k_mlp_epi(float* __restrict__ y, const float* __restrict__ bias, int rows) {
  int lane = threadIdx.x & 31;
  int r = blockIdx.x * (blockDim.x >> 5) + (threadIdx.x >> 5);
  if (r >= rows) return;
  float v = 0.f;
  if (lane < L2F) v = y[r * L2F + lane] + bias[lane];
  float mu = wave_sum32(lane < L2F ? v : 0.f) * (1.f / 24.f);
  float d = (lane < L2F) ? v - mu : 0.f;
  float var = wave_sum32(d * d) * (1.f / 24.f);
  if (lane < L2F) y[r * L2F + lane] = fmaxf(d * rsqrtf(var + LN_EPS), 0.f);
}

// ---------------- CSR build over ei (bucketed by source node) ----------------
__global__ void k_count(const int* __restrict__ ei0, int* __restrict__ cnt, int ne) {
  int i = blockIdx.x * blockDim.x + threadIdx.x;
  if (i < ne) atomicAdd(&cnt[ei0[i]], 1);
}

__global__ void k_scan(const int* __restrict__ cnt, int* __restrict__ startp,
                       int* __restrict__ cursor) {
  __shared__ int s[NNODE];
  int t = threadIdx.x;
  s[t] = cnt[t];
  __syncthreads();
  for (int off = 1; off < NNODE; off <<= 1) {
    int v = (t >= off) ? s[t - off] : 0;
    __syncthreads();
    s[t] += v;
    __syncthreads();
  }
  startp[t + 1] = s[t];          // inclusive -> start[t+1]
  cursor[t] = s[t] - cnt[t];     // exclusive prefix
  if (t == 0) startp[0] = 0;
}

__global__ void k_fill(const int* __restrict__ ei0, int* __restrict__ cursor,
                       int* __restrict__ eord, int ne) {
  int i = blockIdx.x * blockDim.x + threadIdx.x;
  if (i < ne) {
    int p = atomicAdd(&cursor[ei0[i]], 1);
    eord[p] = i;
  }
}

// ---------------- link kernel: sparse C[:,u,v], Amask/Pmask, fused m3 MLP ----------------
__global__ void k_link(const int* __restrict__ pl, const int* __restrict__ ei1,
                       const int* __restrict__ startp, const int* __restrict__ eord,
                       const float* __restrict__ x1, const float* __restrict__ x2,
                       const float* __restrict__ m3w, const float* __restrict__ m3b,
                       float* __restrict__ vout) {
  __shared__ float accs[4][L2F];
  int lane = threadIdx.x & 31, w = threadIdx.x >> 5;
  int j = blockIdx.x * 4 + w;                 // NL_CNT divisible by 4
  int u = pl[j], vtx = pl[NL_CNT + j];
  if (lane < L2F) accs[w][lane] = 0.f;
  __syncthreads();

  int s0 = startp[u], s1 = startp[u + 1];
  bool hit = false;
  for (int t = s0 + lane; t < s1; t += 32) hit |= (ei1[eord[t]] == vtx);
  bool in_ei = __any(hit);

  bool found = false;
  for (int p = s0; p < s1; ++p) {
    int e1 = eord[p];
    int m  = ei1[e1];
    int ms = startp[m], me = startp[m + 1];
    for (int t = ms + lane; t < me; t += 32) {
      int e2 = eord[t];
      if (ei1[e2] == vtx) {
        found = true;
#pragma unroll
        for (int l = 0; l < L2F; ++l)
          atomicAdd(&accs[w][l], x1[e1 * L2F + l] * x2[e2 * L2F + l]);
      }
    }
  }
  bool in_union = __any(found) || in_ei;     // Pmask | in_ei
  __syncthreads();

  float y = 0.f;
  if (lane < L2F) {
    y = m3b[lane];
#pragma unroll 4
    for (int c = 0; c < L2F; ++c) y += accs[w][c] * m3w[c * L2F + lane];
    if (in_ei) y += m3w[L2F * L2F + lane];   // feat[24] = in_ei
  }
  float mu = wave_sum32(lane < L2F ? y : 0.f) * (1.f / 24.f);
  float d = (lane < L2F) ? y - mu : 0.f;
  float var = wave_sum32(d * d) * (1.f / 24.f);
  float r = fmaxf(d * rsqrtf(var + LN_EPS), 0.f);
  if (lane < L2F) vout[j * L2F + lane] = in_union ? r : 0.f;
}

// ---------------- final head ----------------
__global__ void k_final(const float* __restrict__ h, const float* __restrict__ v,
                        const float* __restrict__ m4w, const float* __restrict__ m4b,
                        float* __restrict__ out) {
  int i = blockIdx.x * blockDim.x + threadIdx.x;
  if (i >= EI_CNT) return;
  float z = m4b[0];
#pragma unroll 8
  for (int c = 0; c < 32; ++c)
    z += (h[i * 32 + c] + h[(i + EI_CNT) * 32 + c]) * m4w[c];
#pragma unroll 8
  for (int l = 0; l < 24; ++l)
    z += (v[i * 24 + l] + v[(i + EI_CNT) * 24 + l]) * m4w[32 + l];
  out[i] = 1.f / (1.f + __expf(-z));
}

// ---------------- host orchestration ----------------
extern "C" void kernel_launch(void* const* d_in, const int* in_sizes, int n_in,
                              void* d_out, int out_size, void* d_ws, size_t ws_size,
                              hipStream_t stream) {
  const float* x      = (const float*)d_in[0];
  const int*   edge2  = (const int*)d_in[1];    // [2][E2]
  const int*   edge2r = (const int*)d_in[2];
  const int*   ei     = (const int*)d_in[3];    // [2][EI]
  const int*   plinks = (const int*)d_in[4];    // [2][NL]
  const int*   posm   = (const int*)d_in[6];
  const float* w0  = (const float*)d_in[7];
  const float* b0  = (const float*)d_in[8];
  const float* w1  = (const float*)d_in[9];
  const float* b1  = (const float*)d_in[10];
  const float* wr0 = (const float*)d_in[11];
  const float* br0 = (const float*)d_in[12];
  const float* wr1 = (const float*)d_in[13];
  const float* br1 = (const float*)d_in[14];
  const float* m1w = (const float*)d_in[15];
  const float* m1b = (const float*)d_in[16];
  const float* m2w = (const float*)d_in[17];
  const float* m2b = (const float*)d_in[18];
  const float* m3w = (const float*)d_in[19];
  const float* m3b = (const float*)d_in[20];
  const float* m4w = (const float*)d_in[21];
  const float* m4b = (const float*)d_in[22];
  float* out = (float*)d_out;

  uint8_t* base = (uint8_t*)d_ws;
  size_t off = 0;
  auto alloc = [&](size_t bytes) -> void* {
    off = (off + 255) & ~(size_t)255;
    void* p = base + off;
    off += bytes;
    return p;
  };
  int*       degA  = (int*)alloc(P_TOT * 4);
  int*       degB  = (int*)alloc(P_TOT * 4);
  float*     dinvA = (float*)alloc(P_TOT * 4);
  float*     dinvB = (float*)alloc(P_TOT * 4);
  _Float16*  xh    = (_Float16*)alloc((size_t)P_TOT * NFEAT * 2);
  // packed weights: NT * K32 * 512 halves
  _Float16*  w0p   = (_Float16*)alloc(2 * 4 * 512 * 2);
  _Float16*  wr0p  = (_Float16*)alloc(2 * 4 * 512 * 2);
  _Float16*  w1p   = (_Float16*)alloc(2 * 1 * 512 * 2);
  _Float16*  wr1p  = (_Float16*)alloc(2 * 1 * 512 * 2);
  _Float16*  m1wp  = (_Float16*)alloc(2 * 4 * 512 * 2);
  _Float16*  m2wp  = (_Float16*)alloc(2 * 4 * 512 * 2);
  float*     t0    = (float*)alloc((size_t)P_TOT * 32 * 4);
  float*     tr0   = (float*)alloc((size_t)P_TOT * 32 * 4);
  float*     acc0  = (float*)alloc((size_t)P_TOT * 32 * 4);
  float*     acc1  = (float*)alloc((size_t)P_TOT * 32 * 4);
  float*     h1    = (float*)alloc((size_t)P_TOT * 32 * 4);
  _Float16*  h1h   = (_Float16*)alloc((size_t)P_TOT * 32 * 2);
  float*     h2    = (float*)alloc((size_t)P_TOT * 32 * 4);
  _Float16*  xph   = (_Float16*)alloc((size_t)EI_CNT * NFEAT * 2);
  float*     x1b   = (float*)alloc((size_t)EI_CNT * L2F * 4);
  float*     x2b   = (float*)alloc((size_t)EI_CNT * L2F * 4);
  int*       cnt   = (int*)alloc(NNODE * 4);
  int*       startp= (int*)alloc((NNODE + 1) * 4);
  int*       cursor= (int*)alloc(NNODE * 4);
  int*       eord  = (int*)alloc(EI_CNT * 4);
  float*     vbuf  = (float*)alloc((size_t)NL_CNT * L2F * 4);

  hipMemsetAsync(degA, 0, P_TOT * 4, stream);
  hipMemsetAsync(degB, 0, P_TOT * 4, stream);
  hipMemsetAsync(acc0, 0, (size_t)P_TOT * 32 * 4, stream);
  hipMemsetAsync(acc1, 0, (size_t)P_TOT * 32 * 4, stream);
  hipMemsetAsync(cnt,  0, NNODE * 4, stream);

  // f16 cast of features + packed weight layouts
  k_cast_f16<<<((size_t)P_TOT * NFEAT + 255) / 256, 256, 0, stream>>>(x, xh, P_TOT * NFEAT);
  auto pack = [&](const float* B, _Float16* Bp, int N, int K) {
    int total = ((N + 15) / 16) * (K / 32) * 32;
    k_pack_b<<<(total + 255) / 256, 256, 0, stream>>>(B, Bp, N, K);
  };
  pack(w0, w0p, L1F, NFEAT);
  pack(wr0, wr0p, L1F, NFEAT);
  pack(w1, w1p, L1F, L1F);
  pack(wr1, wr1p, L1F, L1F);
  pack(m1w, m1wp, L2F, NFEAT);
  pack(m2w, m2wp, L2F, NFEAT);

  // degrees + dinv
  k_deg<<<(E2_CNT + 255) / 256, 256, 0, stream>>>(edge2 + E2_CNT, degA, E2_CNT);
  k_deg<<<(E2_CNT + 255) / 256, 256, 0, stream>>>(edge2r + E2_CNT, degB, E2_CNT);
  k_dinv<<<(P_TOT + 255) / 256, 256, 0, stream>>>(degA, dinvA, P_TOT);
  k_dinv<<<(P_TOT + 255) / 256, 256, 0, stream>>>(degB, dinvB, P_TOT);

  // layer 1: WMMA GEMMs + scatter + epilogue
  {
    int tiles = (P_TOT / 16) * 2;
    k_gemm_wmma<NFEAT><<<(tiles + 3) / 4, 128, 0, stream>>>(xh, w0p, t0, P_TOT, L1F);
    k_gemm_wmma<NFEAT><<<(tiles + 3) / 4, 128, 0, stream>>>(xh, wr0p, tr0, P_TOT, L1F);
  }
  k_scatter<<<(E2_CNT * 32) / 256, 256, 0, stream>>>(edge2, edge2 + E2_CNT, dinvA, t0, acc0, E2_CNT);
  k_scatter<<<(E2_CNT * 32) / 256, 256, 0, stream>>>(edge2r, edge2r + E2_CNT, dinvB, tr0, acc1, E2_CNT);
  k_gcn_epi<<<(P_TOT + 7) / 8, 256, 0, stream>>>(acc0, t0, dinvA, b0, acc1, tr0, dinvB, br0,
                                                 h1, h1h, P_TOT);

  // layer 2
  hipMemsetAsync(acc0, 0, (size_t)P_TOT * 32 * 4, stream);
  hipMemsetAsync(acc1, 0, (size_t)P_TOT * 32 * 4, stream);
  {
    int tiles = (P_TOT / 16) * 2;
    k_gemm_wmma<L1F><<<(tiles + 3) / 4, 128, 0, stream>>>(h1h, w1p, t0, P_TOT, L1F);
    k_gemm_wmma<L1F><<<(tiles + 3) / 4, 128, 0, stream>>>(h1h, wr1p, tr0, P_TOT, L1F);
  }
  k_scatter<<<(E2_CNT * 32) / 256, 256, 0, stream>>>(edge2, edge2 + E2_CNT, dinvA, t0, acc0, E2_CNT);
  k_scatter<<<(E2_CNT * 32) / 256, 256, 0, stream>>>(edge2r, edge2r + E2_CNT, dinvB, tr0, acc1, E2_CNT);
  k_gcn_epi<<<(P_TOT + 7) / 8, 256, 0, stream>>>(acc0, t0, dinvA, b1, acc1, tr0, dinvB, br1,
                                                 h2, (_Float16*)nullptr, P_TOT);

  // MLP branch: xp = x[pos_mask]; x1/x2 = relu(ln(xp @ m1w/m2w + b))
  k_gather_cast<<<((size_t)EI_CNT * NFEAT + 255) / 256, 256, 0, stream>>>(x, posm, xph, EI_CNT);
  {
    int tiles = (EI_CNT / 16) * 2;   // N=24 -> 2 col tiles (guarded)
    k_gemm_wmma<NFEAT><<<(tiles + 3) / 4, 128, 0, stream>>>(xph, m1wp, x1b, EI_CNT, L2F);
    k_gemm_wmma<NFEAT><<<(tiles + 3) / 4, 128, 0, stream>>>(xph, m2wp, x2b, EI_CNT, L2F);
  }
  k_mlp_epi<<<(EI_CNT + 7) / 8, 256, 0, stream>>>(x1b, m1b, EI_CNT);
  k_mlp_epi<<<(EI_CNT + 7) / 8, 256, 0, stream>>>(x2b, m2b, EI_CNT);

  // CSR of ei, then sparse link evaluation (replaces 24x 1024^3 dense BMM)
  k_count<<<(EI_CNT + 255) / 256, 256, 0, stream>>>(ei, cnt, EI_CNT);
  k_scan<<<1, NNODE, 0, stream>>>(cnt, startp, cursor);
  k_fill<<<(EI_CNT + 255) / 256, 256, 0, stream>>>(ei, cursor, eord, EI_CNT);
  k_link<<<NL_CNT / 4, 128, 0, stream>>>(plinks, ei + EI_CNT, startp, eord,
                                         x1b, x2b, m3w, m3b, vbuf);

  // final head
  k_final<<<(EI_CNT + 255) / 256, 256, 0, stream>>>(h2, vbuf, m4w, m4b, out);
}
